// mol_encoder_10857677325003
// MI455X (gfx1250) — matrix operand order repaired and checked by reading, not compiled, not run on hardware
//
#include <hip/hip_runtime.h>
#include <hip/hip_bf16.h>

// ---------------------------------------------------------------------------
// Types for CDNA5 WMMA (bf16 in, f32 accumulate)
// ---------------------------------------------------------------------------
typedef __bf16 bf16x16 __attribute__((ext_vector_type(16)));
typedef float  floatx8 __attribute__((ext_vector_type(8)));

union Frag16 {
    bf16x16      v;
    uint4        q[2];
    unsigned int u[8];
};

// f32 -> bf16 (round to nearest even), packed as two bf16 in one uint
__device__ __forceinline__ unsigned short f2bf(float f) {
    unsigned u = __float_as_uint(f);
    u += 0x7FFFu + ((u >> 16) & 1u);
    return (unsigned short)(u >> 16);
}
__device__ __forceinline__ unsigned int pack2(float lo, float hi) {
    return (unsigned int)f2bf(lo) | ((unsigned int)f2bf(hi) << 16);
}

// monotonic float<->uint key for atomicMax on floats
__device__ __forceinline__ unsigned fkey(float f) {
    unsigned u = __float_as_uint(f);
    return (u & 0x80000000u) ? ~u : (u | 0x80000000u);
}
__device__ __forceinline__ float funkey(unsigned k) {
    unsigned u = (k & 0x80000000u) ? (k & 0x7FFFFFFFu) : ~k;
    return __uint_as_float(u);
}

// ---------------------------------------------------------------------------
// GEMM: C[M,N] = act( A[M,K] @ B + bias ), bf16 WMMA, f32 accum.
//   transB==0: B is [K,N] row-major.  transB==1: B is [N,K] (compute A@B^T)
//   flags bit0: ReLU epilogue, bit1: accumulate into existing C
// Block: 256 threads (8 waves = 4(M) x 2(N)), tile 128(M) x 64(N), K step 32.
// Each wave: 32x32 output = 2 A-frags x 2 B-frags = 4 WMMA per K step.
// LDS double-buffered: one barrier per K step; next tile staged before
// current tile is consumed (global-load latency hidden behind WMMA).
// NOTE: requires K even (f2 loads 8B-aligned) -- true for all uses here.
// ---------------------------------------------------------------------------
__device__ __forceinline__ void stage_tiles(
    unsigned int (*__restrict__ As)[16], unsigned int (*__restrict__ Bs)[32],
    const float* __restrict__ A, const float* __restrict__ Bm,
    int M, int N, int K, int transB, int bm, int bn, int kk, int tid)
{
    // ---- A tile 128x32 ----
    if (bm + 128 <= M && kk + 32 <= K) {
        #pragma unroll
        for (int it = 0; it < 8; ++it) {
            int idx = tid + it * 256;           // 2048 uints
            int r = idx >> 4, cp = idx & 15;
            float2 v = *(const float2*)&A[(long)(bm + r) * K + kk + 2 * cp];
            As[r][cp] = pack2(v.x, v.y);
        }
    } else {
        #pragma unroll
        for (int it = 0; it < 8; ++it) {
            int idx = tid + it * 256;
            int r = idx >> 4, cp = idx & 15;
            int gr = bm + r, gk = kk + 2 * cp;
            float2 v = make_float2(0.f, 0.f);
            if (gr < M && gk < K) v = *(const float2*)&A[(long)gr * K + gk]; // K even
            As[r][cp] = pack2(v.x, v.y);
        }
    }
    // ---- B tile 32x64 ----
    if (!transB) {
        if (kk + 32 <= K && bn + 64 <= N) {
            #pragma unroll
            for (int it = 0; it < 4; ++it) {
                int idx = tid + it * 256;       // 1024 uints
                int r = idx >> 5, cp = idx & 31;
                float2 v = *(const float2*)&Bm[(long)(kk + r) * N + bn + 2 * cp];
                Bs[r][cp] = pack2(v.x, v.y);
            }
        } else {
            #pragma unroll
            for (int it = 0; it < 4; ++it) {
                int idx = tid + it * 256;
                int r = idx >> 5, cp = idx & 31;
                int gk = kk + r, gn = bn + 2 * cp;
                float2 v = make_float2(0.f, 0.f);
                if (gk < K && gn < N) v = *(const float2*)&Bm[(long)gk * N + gn]; // N even
                Bs[r][cp] = pack2(v.x, v.y);
            }
        }
    } else {
        if (kk + 32 <= K && bn + 64 <= N) {
            #pragma unroll
            for (int it = 0; it < 4; ++it) {
                int idx = tid + it * 256;
                int r = idx >> 5, cp = idx & 31;
                int gk = kk + r, gn = bn + 2 * cp;
                float f0 = Bm[(long)gn * K + gk];
                float f1 = Bm[(long)(gn + 1) * K + gk];
                Bs[r][cp] = pack2(f0, f1);
            }
        } else {
            #pragma unroll
            for (int it = 0; it < 4; ++it) {
                int idx = tid + it * 256;
                int r = idx >> 5, cp = idx & 31;
                int gk = kk + r, gn = bn + 2 * cp;
                float f0 = 0.f, f1 = 0.f;
                if (gk < K && gn < N) {                 // N even -> gn+1 < N too
                    f0 = Bm[(long)gn * K + gk];
                    f1 = Bm[(long)(gn + 1) * K + gk];
                }
                Bs[r][cp] = pack2(f0, f1);
            }
        }
    }
}

__device__ __forceinline__ void store_c(float* __restrict__ C, const float* __restrict__ bias,
                                        int M, int N, int flags, int gm, int gn, float acc)
{
    if (gm < M && gn < N) {
        long idx = (long)gm * N + gn;
        float v = acc + (bias ? bias[gn] : 0.f);
        if (flags & 2) v += C[idx];
        if (flags & 1) v = fmaxf(v, 0.f);
        C[idx] = v;
    }
}

__global__ __launch_bounds__(256)
void gemm_bf16_wmma(const float* __restrict__ A, const float* __restrict__ Bm,
                    const float* __restrict__ bias, float* __restrict__ C,
                    int M, int N, int K, int transB, int flags)
{
    __shared__ __align__(16) unsigned int As[2][128][16]; // 2 x 8 KB
    __shared__ __align__(16) unsigned int Bs[2][32][32];  // 2 x 4 KB

    const int tid   = threadIdx.x;
    const int lane  = tid & 31;
    const int wv    = tid >> 5;       // 0..7
    const int wm    = wv & 3;         // M sub-tile (32 rows each)
    const int wn    = wv >> 2;        // N sub-tile (32 cols each)
    const int l15   = lane & 15;
    const int khalf = lane >> 4;      // 0/1
    const int bm    = blockIdx.y * 128;
    const int bn    = blockIdx.x * 64;

    floatx8 acc00 = {}, acc01 = {}, acc10 = {}, acc11 = {};

    stage_tiles(As[0], Bs[0], A, Bm, M, N, K, transB, bm, bn, 0, tid);

    int cur = 0;
    for (int kk = 0; kk < K; kk += 32, cur ^= 1) {
        __syncthreads();   // staging of As[cur]/Bs[cur] visible to all waves

        // stage next tile into the other buffer (overlaps with WMMA below)
        if (kk + 32 < K)
            stage_tiles(As[cur ^ 1], Bs[cur ^ 1], A, Bm, M, N, K, transB,
                        bm, bn, kk + 32, tid);
        // speculative prefetch two steps ahead (global_prefetch_b8)
        if (kk + 64 < K && tid < 128) {
            int gr = bm + tid;
            if (gr < M) __builtin_prefetch(&A[(long)gr * K + kk + 64], 0, 1);
        }

        // ---- fragments from current buffer ----
        // A: lane 0-15 rows, khalf selects K {0-7,16-23} vs {8-15,24-31}
        Frag16 fa0, fa1, fb0, fb1;
        int ar0 = wm * 32 + l15;
        int ar1 = ar0 + 16;
        fa0.q[0] = *(const uint4*)&As[cur][ar0][khalf * 4];
        fa0.q[1] = *(const uint4*)&As[cur][ar0][8 + khalf * 4];
        fa1.q[0] = *(const uint4*)&As[cur][ar1][khalf * 4];
        fa1.q[1] = *(const uint4*)&As[cur][ar1][8 + khalf * 4];
        // B: lane holds N col (wn*32 + l15 [+16]), K {0-7,16-23} vs {8-15,24-31} by khalf
        int bnc = wn * 32 + l15;
        int bc = bnc >> 1;
        int bsh = (bnc & 1) * 16;
        #pragma unroll
        for (int j = 0; j < 8; ++j) {
            int bk = 2 * j + 8 * khalf + 2 * (j & 4);
            fb0.u[j] = ((Bs[cur][bk][bc] >> bsh) & 0xFFFFu) | (((Bs[cur][bk + 1][bc] >> bsh) & 0xFFFFu) << 16);
            fb1.u[j] = ((Bs[cur][bk][bc + 8] >> bsh) & 0xFFFFu) | (((Bs[cur][bk + 1][bc + 8] >> bsh) & 0xFFFFu) << 16);
        }

        acc00 = __builtin_amdgcn_wmma_f32_16x16x32_bf16(false, fa0.v, false, fb0.v,
                                                        (short)0, acc00, false, false);
        acc01 = __builtin_amdgcn_wmma_f32_16x16x32_bf16(false, fa0.v, false, fb1.v,
                                                        (short)0, acc01, false, false);
        acc10 = __builtin_amdgcn_wmma_f32_16x16x32_bf16(false, fa1.v, false, fb0.v,
                                                        (short)0, acc10, false, false);
        acc11 = __builtin_amdgcn_wmma_f32_16x16x32_bf16(false, fa1.v, false, fb1.v,
                                                        (short)0, acc11, false, false);
    }

    // C layout: VGPR r -> M = r + 8*khalf, N = lane&15
    const int gm0 = bm + wm * 32 + khalf * 8;
    const int gn0 = bn + wn * 32 + l15;
    #pragma unroll
    for (int r = 0; r < 8; ++r) {
        store_c(C, bias, M, N, flags, gm0 + r,      gn0,      acc00[r]);
        store_c(C, bias, M, N, flags, gm0 + r,      gn0 + 16, acc01[r]);
        store_c(C, bias, M, N, flags, gm0 + r + 16, gn0,      acc10[r]);
        store_c(C, bias, M, N, flags, gm0 + r + 16, gn0 + 16, acc11[r]);
    }
}

// ---------------------------------------------------------------------------
// GAT attention kernels (H=2, C=256 fixed)
// ---------------------------------------------------------------------------
__global__ void attn_coeff(const float* __restrict__ xh,
                           const float* __restrict__ a_s, const float* __restrict__ a_d,
                           float* __restrict__ asrc, float* __restrict__ adst, int Nn)
{
    long w = ((long)blockIdx.x * blockDim.x + threadIdx.x) >> 5;
    int lane = threadIdx.x & 31;
    if (w >= Nn) return;
    #pragma unroll
    for (int h = 0; h < 2; ++h) {
        const float* row = xh + w * 512 + h * 256;
        float s1 = 0.f, s2 = 0.f;
        for (int c = lane; c < 256; c += 32) {
            float v = row[c];
            s1 += v * a_s[h * 256 + c];
            s2 += v * a_d[h * 256 + c];
        }
        #pragma unroll
        for (int off = 16; off > 0; off >>= 1) {
            s1 += __shfl_xor(s1, off, 32);
            s2 += __shfl_xor(s2, off, 32);
        }
        if (lane == 0) { asrc[w * 2 + h] = s1; adst[w * 2 + h] = s2; }
    }
}

__device__ __forceinline__ void edge_sd(const int* ei, long i, int E, int* s, int* d) {
    *s = (i < E) ? ei[i]     : (int)(i - E);  // self loops appended
    *d = (i < E) ? ei[E + i] : (int)(i - E);
}

__global__ void attn_max(const int* __restrict__ ei, const float* __restrict__ asrc,
                         const float* __restrict__ adst, unsigned* __restrict__ mk,
                         int E, int Nn)
{
    long i = (long)blockIdx.x * blockDim.x + threadIdx.x;
    if (i >= (long)E + Nn) return;
    int s, d; edge_sd(ei, i, E, &s, &d);
    #pragma unroll
    for (int h = 0; h < 2; ++h) {
        float e = asrc[s * 2 + h] + adst[d * 2 + h];
        e = e > 0.f ? e : 0.2f * e;
        atomicMax(&mk[d * 2 + h], fkey(e));
    }
}

__global__ void attn_sum(const int* __restrict__ ei, const float* __restrict__ asrc,
                         const float* __restrict__ adst, const unsigned* __restrict__ mk,
                         float* __restrict__ ss, int E, int Nn)
{
    long i = (long)blockIdx.x * blockDim.x + threadIdx.x;
    if (i >= (long)E + Nn) return;
    int s, d; edge_sd(ei, i, E, &s, &d);
    #pragma unroll
    for (int h = 0; h < 2; ++h) {
        float e = asrc[s * 2 + h] + adst[d * 2 + h];
        e = e > 0.f ? e : 0.2f * e;
        atomicAdd(&ss[d * 2 + h], __expf(e - funkey(mk[d * 2 + h])));
    }
}

// head-mean folded in as 0.5 factor (mean is linear); float4 gathers
__global__ void attn_scatter(const int* __restrict__ ei, const float* __restrict__ asrc,
                             const float* __restrict__ adst, const unsigned* __restrict__ mk,
                             const float* __restrict__ ss, const float* __restrict__ xh,
                             float* __restrict__ out, int E, int Nn)
{
    long w = ((long)blockIdx.x * blockDim.x + threadIdx.x) >> 5;
    int lane = threadIdx.x & 31;
    if (w >= (long)E + Nn) return;
    int s, d; edge_sd(ei, w, E, &s, &d);
    float al0, al1;
    {
        float e0 = asrc[s * 2 + 0] + adst[d * 2 + 0];
        e0 = e0 > 0.f ? e0 : 0.2f * e0;
        al0 = 0.5f * __expf(e0 - funkey(mk[d * 2 + 0])) / ss[d * 2 + 0];
        float e1 = asrc[s * 2 + 1] + adst[d * 2 + 1];
        e1 = e1 > 0.f ? e1 : 0.2f * e1;
        al1 = 0.5f * __expf(e1 - funkey(mk[d * 2 + 1])) / ss[d * 2 + 1];
    }
    const float4* x4 = (const float4*)(xh + (long)s * 512); // head0: [0,64), head1: [64,128)
    float* o = out + (long)d * 256;
    #pragma unroll
    for (int it = 0; it < 2; ++it) {
        int j = lane + it * 32;                  // 64 float4 = 256 floats
        float4 a = x4[j];
        float4 b = x4[64 + j];
        atomicAdd(&o[4 * j + 0], al0 * a.x + al1 * b.x);
        atomicAdd(&o[4 * j + 1], al0 * a.y + al1 * b.y);
        atomicAdd(&o[4 * j + 2], al0 * a.z + al1 * b.z);
        atomicAdd(&o[4 * j + 3], al0 * a.w + al1 * b.w);
    }
}

// ---------------------------------------------------------------------------
// Elementwise / segment kernels
// ---------------------------------------------------------------------------
__global__ void bias_act_kernel(float* __restrict__ x, const float* __restrict__ b,
                                long n, int C, int relu)
{
    long i = (long)blockIdx.x * blockDim.x + threadIdx.x;
    if (i >= n) return;
    float v = x[i] + (b ? b[(int)(i % C)] : 0.f);
    if (relu) v = fmaxf(v, 0.f);
    x[i] = v;
}

__global__ void gate_kernel(const float* __restrict__ g, const float* __restrict__ molb,
                            const float* __restrict__ x, float* __restrict__ h,
                            long n, int C)
{
    long i = (long)blockIdx.x * blockDim.x + threadIdx.x;
    if (i >= n) return;
    float z = 1.f / (1.f + __expf(-(g[i] + molb[(int)(i % C)])));
    h[i] = z * x[i] + (1.f - z) * h[i];
}

__global__ void concat_hx_kernel(const float* __restrict__ h, const float* __restrict__ molx,
                                 float* __restrict__ hx, int Nn)
{
    long i = (long)blockIdx.x * blockDim.x + threadIdx.x;
    if (i >= (long)Nn * 334) return;
    int n = (int)(i / 334), j = (int)(i % 334);
    hx[i] = (j < 256) ? h[(long)n * 256 + j] : molx[(long)n * 78 + (j - 256)];
}

__global__ void degrees_kernel(const int* __restrict__ mei, float* __restrict__ D,
                               float* __restrict__ B, int E)
{
    long i = (long)blockIdx.x * blockDim.x + threadIdx.x;
    if (i >= E) return;
    atomicAdd(&D[mei[i]], 1.f);      // node index
    atomicAdd(&B[mei[E + i]], 1.f);  // hyperedge index
}

// one wave per edge, C==128: one float4 per lane
__global__ void seg_scatter_kernel(const int* __restrict__ gidx, const int* __restrict__ sidx,
                                   const float* __restrict__ in, float* __restrict__ out,
                                   int E)
{
    long w = ((long)blockIdx.x * blockDim.x + threadIdx.x) >> 5;
    int lane = threadIdx.x & 31;
    if (w >= E) return;
    const float4* src = (const float4*)(in + (long)gidx[w] * 128);
    float* dst = out + (long)sidx[w] * 128;
    float4 a = src[lane];
    atomicAdd(&dst[4 * lane + 0], a.x);
    atomicAdd(&dst[4 * lane + 1], a.y);
    atomicAdd(&dst[4 * lane + 2], a.z);
    atomicAdd(&dst[4 * lane + 3], a.w);
}

__global__ void degscale_kernel(float* __restrict__ x, const float* __restrict__ deg,
                                const float* __restrict__ b, long n, int C, int relu)
{
    long i = (long)blockIdx.x * blockDim.x + threadIdx.x;
    if (i >= n) return;
    int node = (int)(i / C), c = (int)(i % C);
    float dv = deg[node];
    float v = x[i] * (dv > 0.f ? 1.f / dv : 0.f) + (b ? b[c] : 0.f);
    if (relu) v = fmaxf(v, 0.f);
    x[i] = v;
}

__global__ void out_kernel(const float* __restrict__ h, const float* __restrict__ hx2,
                           float* __restrict__ out, int Nn)
{
    long i = (long)blockIdx.x * blockDim.x + threadIdx.x;
    if (i >= (long)Nn * 384) return;
    int n = (int)(i / 384), j = (int)(i % 384);
    out[i] = (j < 256) ? h[(long)n * 256 + j] : hx2[(long)n * 128 + (j - 256)];
}

// ---------------------------------------------------------------------------
// Host orchestration
// ---------------------------------------------------------------------------
static inline void launch_gemm(const float* A, const float* B, const float* bias, float* C,
                               int M, int Nc, int K, int transB, int flags, hipStream_t st)
{
    dim3 grid((Nc + 63) / 64, (M + 127) / 128);
    gemm_bf16_wmma<<<grid, 256, 0, st>>>(A, B, bias, C, M, Nc, K, transB, flags);
}

static inline long gblk(long n) { return (n + 255) / 256; }

static void run_gat(const float* xin, int Kin, const int* mei, int E, int Nn,
                    const float* W, const float* a_s, const float* a_d, const float* b,
                    float* xh, float* asrc, float* adst, unsigned* mk, float* ss,
                    float* outbuf, int relu, hipStream_t st)
{
    launch_gemm(xin, W, nullptr, xh, Nn, 512, Kin, 0, 0, st);
    attn_coeff<<<gblk((long)Nn * 32), 256, 0, st>>>(xh, a_s, a_d, asrc, adst, Nn);
    hipMemsetAsync(mk, 0, (size_t)Nn * 2 * sizeof(unsigned), st);
    hipMemsetAsync(ss, 0, (size_t)Nn * 2 * sizeof(float), st);
    long E2 = (long)E + Nn;
    attn_max<<<gblk(E2), 256, 0, st>>>(mei, asrc, adst, mk, E, Nn);
    attn_sum<<<gblk(E2), 256, 0, st>>>(mei, asrc, adst, mk, ss, E, Nn);
    hipMemsetAsync(outbuf, 0, (size_t)Nn * 256 * sizeof(float), st);
    attn_scatter<<<gblk(E2 * 32), 256, 0, st>>>(mei, asrc, adst, mk, ss, xh, outbuf, E, Nn);
    bias_act_kernel<<<gblk((long)Nn * 256), 256, 0, st>>>(outbuf, b, (long)Nn * 256, 256, relu);
}

extern "C" void kernel_launch(void* const* d_in, const int* in_sizes, int n_in,
                              void* d_out, int out_size, void* d_ws, size_t ws_size,
                              hipStream_t stream)
{
    (void)n_in; (void)out_size; (void)ws_size;
    const int Nn = in_sizes[0] / 78;
    const int E  = in_sizes[1] / 2;

    const float* mol_x   = (const float*)d_in[0];
    const int*   mei     = (const int*)d_in[1];   // [2,E]; row0=src/node, row1=dst/edge
    const float* W1      = (const float*)d_in[3];
    const float* as1     = (const float*)d_in[4];
    const float* ad1     = (const float*)d_in[5];
    const float* b1      = (const float*)d_in[6];
    const float* W2      = (const float*)d_in[7];
    const float* as2     = (const float*)d_in[8];
    const float* ad2     = (const float*)d_in[9];
    const float* b2      = (const float*)d_in[10];
    const float* W3      = (const float*)d_in[11];
    const float* as3     = (const float*)d_in[12];
    const float* ad3     = (const float*)d_in[13];
    const float* b3      = (const float*)d_in[14];
    const float* fc1_w   = (const float*)d_in[15];
    const float* fc1_b   = (const float*)d_in[16];
    const float* fc2_w   = (const float*)d_in[17];
    const float* fc2_b   = (const float*)d_in[18];
    const float* mol_b   = (const float*)d_in[19];
    const float* theta1  = (const float*)d_in[20];
    const float* hb1     = (const float*)d_in[21];
    const float* theta2  = (const float*)d_in[22];
    const float* hb2     = (const float*)d_in[23];
    float* out = (float*)d_out;

    // workspace layout (floats), hyper-phase overlays the gate/x regions
    float* ws = (float*)d_ws;
    size_t fN = (size_t)Nn;
    float*    xh   = ws;                         // N*512
    float*    h    = xh  + fN * 512;             // N*256
    float*    xb   = h   + fN * 256;             // N*256
    float*    g    = xb  + fN * 256;             // N*256
    float*    asrc = g   + fN * 256;             // N*2
    float*    adst = asrc + fN * 2;              // N*2
    unsigned* mk   = (unsigned*)(adst + fN * 2); // N*2
    float*    ss   = (float*)mk + fN * 2;        // N*2
    // hyper phase overlays (xb/g dead by then)
    float* hx   = xh;              // N*334  (<= N*512)
    float* xt   = xb;              // N*128
    float* ef   = xb + fN * 128;   // N*128
    float* nf   = g;               // N*128
    float* Ddeg = g + fN * 128;    // N
    float* Bdeg = Ddeg + fN;       // N

    const int* node_idx = mei;      // mol_edge_index[0]
    const int* edge_idx = mei + E;  // mol_edge_index[1]
    long nC = (long)Nn * 256, nH = (long)Nn * 128;

    // ---- GAT layer 1: h = relu(gat(mol_x)) ----
    run_gat(mol_x, 78, mei, E, Nn, W1, as1, ad1, b1, xh, asrc, adst, mk, ss, h, 1, stream);
    // ---- GAT layer 2: x = relu(gat(h)); gated fuse into h ----
    run_gat(h, 256, mei, E, Nn, W2, as2, ad2, b2, xh, asrc, adst, mk, ss, xb, 1, stream);
    launch_gemm(xb, fc1_w, fc1_b, g, Nn, 256, 256, 1, 0, stream);
    launch_gemm(h,  fc2_w, fc2_b, g, Nn, 256, 256, 1, 2, stream);
    gate_kernel<<<gblk(nC), 256, 0, stream>>>(g, mol_b, xb, h, nC, 256);
    // ---- GAT layer 3 (no relu); gated fuse into h ----
    run_gat(h, 256, mei, E, Nn, W3, as3, ad3, b3, xh, asrc, adst, mk, ss, xb, 0, stream);
    launch_gemm(xb, fc1_w, fc1_b, g, Nn, 256, 256, 1, 0, stream);
    launch_gemm(h,  fc2_w, fc2_b, g, Nn, 256, 256, 1, 2, stream);
    gate_kernel<<<gblk(nC), 256, 0, stream>>>(g, mol_b, xb, h, nC, 256);

    // ---- Hypergraph branch ----
    concat_hx_kernel<<<gblk((long)Nn * 334), 256, 0, stream>>>(h, mol_x, hx, Nn);
    hipMemsetAsync(Ddeg, 0, fN * sizeof(float), stream);
    hipMemsetAsync(Bdeg, 0, fN * sizeof(float), stream);
    degrees_kernel<<<gblk(E), 256, 0, stream>>>(mei, Ddeg, Bdeg, E);

    // layer 1: xt = hx@theta1; ef = Binv*seg(xt[node]->edge); nf = relu(Dinv*seg(ef[edge]->node)+hb1)
    launch_gemm(hx, theta1, nullptr, xt, Nn, 128, 334, 0, 0, stream);
    hipMemsetAsync(ef, 0, fN * 128 * sizeof(float), stream);
    seg_scatter_kernel<<<gblk((long)E * 32), 256, 0, stream>>>(node_idx, edge_idx, xt, ef, E);
    degscale_kernel<<<gblk(nH), 256, 0, stream>>>(ef, Bdeg, nullptr, nH, 128, 0);
    hipMemsetAsync(nf, 0, fN * 128 * sizeof(float), stream);
    seg_scatter_kernel<<<gblk((long)E * 32), 256, 0, stream>>>(edge_idx, node_idx, ef, nf, E);
    degscale_kernel<<<gblk(nH), 256, 0, stream>>>(nf, Ddeg, hb1, nH, 128, 1);

    // layer 2 (nf is input; xt/ef reused; result overwrites nf)
    launch_gemm(nf, theta2, nullptr, xt, Nn, 128, 128, 0, 0, stream);
    hipMemsetAsync(ef, 0, fN * 128 * sizeof(float), stream);
    seg_scatter_kernel<<<gblk((long)E * 32), 256, 0, stream>>>(node_idx, edge_idx, xt, ef, E);
    degscale_kernel<<<gblk(nH), 256, 0, stream>>>(ef, Bdeg, nullptr, nH, 128, 0);
    hipMemsetAsync(nf, 0, fN * 128 * sizeof(float), stream);
    seg_scatter_kernel<<<gblk((long)E * 32), 256, 0, stream>>>(edge_idx, node_idx, ef, nf, E);
    degscale_kernel<<<gblk(nH), 256, 0, stream>>>(nf, Ddeg, hb2, nH, 128, 1);

    // ---- final concat [h | hx2] -> out[N,384] ----
    out_kernel<<<gblk((long)Nn * 384), 256, 0, stream>>>(h, nf, out, Nn);
}
